// TauMamba_10230612099834
// MI455X (gfx1250) — compile-verified, matrix operand order checked by reading, not run on hardware
//
#include <hip/hip_runtime.h>

// ---------------------------------------------------------------------------
// TauMamba single-step for MI455X (gfx1250, wave32, WMMA bf16)
// GEMMs: v_wmma_f32_16x16x32_bf16, B tile staged in LDS via
// global_load_async_to_lds_b128 (ASYNCcnt), double-buffered; A fragments
// direct from global (+prefetch).
// ---------------------------------------------------------------------------
#define B_SZ     512
#define IN_DIM   512
#define HID      1024
#define CLIP     512
#define HEADDIM  64
#define D_STATE  64
#define D_CONV   4
#define NGROUPS  4
#define D_INNER  2048                       // EXPAND * HID
#define NHEADS   32                         // D_INNER / HEADDIM
#define CONV_DIM (D_INNER + 2*NGROUPS*D_STATE)       // 2560
#define D_IN_PROJ (2*D_INNER + 2*NGROUPS*D_STATE + NHEADS) // 4640
#define EPS      1e-5f

typedef __bf16 v16bf __attribute__((ext_vector_type(16)));
typedef __bf16 v8bf  __attribute__((ext_vector_type(8)));
typedef float  v8f   __attribute__((ext_vector_type(8)));
typedef int    v4i_  __attribute__((ext_vector_type(4)));
typedef unsigned short ushort_t;

#if __has_builtin(__builtin_amdgcn_global_load_async_to_lds_b128)
#define USE_ASYNC_LDS 1
#else
#define USE_ASYNC_LDS 0
#endif

// ---- f32 -> bf16 round-to-nearest-even ------------------------------------
__device__ __forceinline__ ushort_t f2bf(float f) {
    unsigned int u = __float_as_uint(f);
    u += 0x7FFFu + ((u >> 16) & 1u);
    return (ushort_t)(u >> 16);
}
__device__ __forceinline__ float sigmoidf_(float x) { return 1.0f / (1.0f + __expf(-x)); }
__device__ __forceinline__ float siluf_(float x)    { return x * sigmoidf_(x); }

// ---------------------------------------------------------------------------
// Weight conversion f32 -> bf16 (ushort storage)
// ---------------------------------------------------------------------------
__global__ void cvt_bf16_kernel(const float* __restrict__ src,
                                ushort_t* __restrict__ dst, int n) {
    int i = blockIdx.x * blockDim.x + threadIdx.x;
    if (i < n) dst[i] = f2bf(src[i]);
}

// ---------------------------------------------------------------------------
// LayerNorm of frame_feat -> x (bf16), one block per row of 512
// ---------------------------------------------------------------------------
__global__ void prenorm_kernel(const float* __restrict__ x,
                               const float* __restrict__ w,
                               const float* __restrict__ b,
                               ushort_t* __restrict__ out) {
    __shared__ float s1[256], s2[256];
    int row = blockIdx.x, t = threadIdx.x;
    const float* xr = x + (size_t)row * IN_DIM;
    float v0 = xr[t], v1 = xr[t + 256];
    s1[t] = v0 + v1;
    s2[t] = v0 * v0 + v1 * v1;
    __syncthreads();
    for (int s = 128; s > 0; s >>= 1) {
        if (t < s) { s1[t] += s1[t + s]; s2[t] += s2[t + s]; }
        __syncthreads();
    }
    float mu  = s1[0] * (1.0f / IN_DIM);
    float var = s2[0] * (1.0f / IN_DIM) - mu * mu;
    float inv = rsqrtf(var + EPS);
    ushort_t* o = out + (size_t)row * IN_DIM;
    o[t]       = f2bf((v0 - mu) * inv * w[t]       + b[t]);
    o[t + 256] = f2bf((v1 - mu) * inv * w[t + 256] + b[t + 256]);
}

// ---------------------------------------------------------------------------
// Generic WMMA GEMM:  C[M,N](f32) = A[M,K](bf16) @ W[N,K](bf16)^T (+ bias)
// Block = 256 threads = 8 waves; wave computes 16(M) x 64(N); block 128 x 64.
// B tile (64 cols x 32 K, 4 KB) staged in LDS once per block per K-step,
// double-buffered; async-to-LDS path (ASYNCcnt) when the toolchain has it.
// M must be a multiple of 128, K a multiple of 32; N arbitrary (guarded).
// ---------------------------------------------------------------------------
__global__ void gemm_bf16_wmma_kernel(const ushort_t* __restrict__ A,
                                      const ushort_t* __restrict__ W,
                                      const float* __restrict__ bias,
                                      float* __restrict__ C,
                                      int N, int K) {
    __shared__ ushort_t Bs[2][64 * 32];          // double-buffered B tile (8 KB)

    const int lane  = threadIdx.x & 31;
    const int wave  = threadIdx.x >> 5;
    const int row0  = blockIdx.y * 128 + wave * 16;
    const int col0  = blockIdx.x * 64;
    const int nlane = lane & 15;
    const int kg    = (lane >> 4) << 3;          // 0 or 8 (half-wave K-group)
    const int mrow  = row0 + nlane;

    // Cooperative B staging: 256 threads move 64 rows x 64 B (one b128 each).
    const int srow   = threadIdx.x >> 2;         // 0..63  (weight row / out col)
    const int schunk = (threadIdx.x & 3) << 3;   // 0,8,16,24 (k element offset)
    int scol = col0 + srow;
    if (scol >= N) scol = N - 1;                 // clamp: garbage cols unstored
    const ushort_t* sgl = W + (size_t)scol * K + schunk;

    const ushort_t* arow = A + (size_t)mrow * K;
    v8f acc[4] = {};                             // 4 x (16x16 f32) accumulators

#define STAGE_B(buf, k0)                                                      \
    do {                                                                      \
        ushort_t* _d = &Bs[buf][srow * 32 + schunk];                          \
        const ushort_t* _s = sgl + (k0);                                      \
        if (USE_ASYNC_LDS) {                                                  \
            __builtin_amdgcn_global_load_async_to_lds_b128(                   \
                (__attribute__((address_space(1))) v4i_*)_s,                  \
                (__attribute__((address_space(3))) v4i_*)_d, 0, 0);           \
        } else {                                                              \
            *(v8bf*)_d = *(const v8bf*)_s;                                    \
        }                                                                     \
    } while (0)

#if USE_ASYNC_LDS
#if __has_builtin(__builtin_amdgcn_s_wait_asynccnt)
#define STAGE_WAIT() __builtin_amdgcn_s_wait_asynccnt(0)
#else
#define STAGE_WAIT() asm volatile("s_wait_asynccnt 0x0" ::: "memory")
#endif
#else
#define STAGE_WAIT() ((void)0)
#endif

    STAGE_B(0, 0);
    STAGE_WAIT();
    __syncthreads();

    for (int k0 = 0; k0 < K; k0 += 32) {
        const int cur = (k0 >> 5) & 1;
        if (k0 + 32 < K) STAGE_B(cur ^ 1, k0 + 32);   // uniform branch

        // A fragment: 16 bf16 per lane, layout per ISA 7.12.2
        v16bf afrag;
        {
            const ushort_t* p = arow + k0 + kg;
            ((v8bf*)&afrag)[0] = *(const v8bf*)(p);
            ((v8bf*)&afrag)[1] = *(const v8bf*)(p + 16);
            __builtin_prefetch(p + 32, 0, 3);         // global_prefetch_b8
        }
#pragma unroll
        for (int j = 0; j < 4; ++j) {
            const ushort_t* q = &Bs[cur][(j * 16 + nlane) * 32 + kg];
            v16bf bfrag;
            ((v8bf*)&bfrag)[0] = *(const v8bf*)(q);
            ((v8bf*)&bfrag)[1] = *(const v8bf*)(q + 16);
            acc[j] = __builtin_amdgcn_wmma_f32_16x16x32_bf16(
                false, afrag, false, bfrag, (short)0, acc[j], false, false);
        }
        STAGE_WAIT();            // next-buffer staging landed in LDS
        __syncthreads();         // all waves done reading current buffer
    }
#undef STAGE_B
#undef STAGE_WAIT

    const int mb = (lane >> 4) << 3;          // 0 or 8 (C row base per half-wave)
#pragma unroll
    for (int j = 0; j < 4; ++j) {
        int col = col0 + j * 16 + nlane;
        if (col < N) {
            float bv = bias ? bias[col] : 0.0f;
#pragma unroll
            for (int v = 0; v < 8; ++v)
                C[(size_t)(row0 + mb + v) * N + col] = acc[j][v] + bv;
        }
    }
}

// ---------------------------------------------------------------------------
// gated = g2 * sigmoid(g1)  (f32 + bf16 copies)
// ---------------------------------------------------------------------------
__global__ void ew_gated_kernel(const float* __restrict__ g1,
                                const float* __restrict__ g2,
                                float* __restrict__ gated,
                                ushort_t* __restrict__ gated_bf) {
    int i = blockIdx.x * blockDim.x + threadIdx.x;
    float v = g2[i] * sigmoidf_(g1[i]);
    gated[i] = v;
    gated_bf[i] = f2bf(v);
}

// ---------------------------------------------------------------------------
// Depthwise conv(4) state roll + SiLU.  One thread per (b, c).
// ---------------------------------------------------------------------------
__global__ void conv_kernel(const float* __restrict__ conv_in,
                            const float* __restrict__ zx,
                            const float* __restrict__ conv_w,
                            const float* __restrict__ conv_b,
                            float* __restrict__ conv_out,
                            float* __restrict__ xBC) {
    int i = blockIdx.x * blockDim.x + threadIdx.x;     // < B_SZ * CONV_DIM
    int b = i / CONV_DIM, c = i - b * CONV_DIM;
    const float* cs = conv_in + (size_t)i * D_CONV;
    float s0 = cs[1], s1 = cs[2], s2 = cs[3];
    float s3 = zx[(size_t)b * D_IN_PROJ + D_INNER + c];
    float* co = conv_out + (size_t)i * D_CONV;
    co[0] = s0; co[1] = s1; co[2] = s2; co[3] = s3;
    const float* w = conv_w + (size_t)c * D_CONV;
    float acc = s0 * w[0] + s1 * w[1] + s2 * w[2] + s3 * w[3] + conv_b[c];
    xBC[i] = siluf_(acc);
}

// ---------------------------------------------------------------------------
// SSM step: one block per (h, b), 64 threads (thread = p).
// Streams state: 1 read + 1 write per element (HBM-bound floor ~23us).
// ---------------------------------------------------------------------------
__global__ void ssm_kernel(const float* __restrict__ ssm_in,
                           const float* __restrict__ zx,
                           const float* __restrict__ xBC,
                           const float* __restrict__ tau,
                           const float* __restrict__ dt_bias,
                           const float* __restrict__ A_log,
                           const float* __restrict__ Dvec,
                           float* __restrict__ ssm_out,
                           float* __restrict__ y) {
    __shared__ float Bs[D_STATE], Cs[D_STATE];
    const int h = blockIdx.x, b = blockIdx.y, p = threadIdx.x;
    const int g = h / (NHEADS / NGROUPS);

    float dtr = zx[(size_t)b * D_IN_PROJ + D_INNER + CONV_DIM + h] + dt_bias[h];
    float dt  = (dtr > 20.0f) ? dtr : log1pf(__expf(dtr));       // softplus
    float Ah  = -__expf(A_log[h]) * tau[b];
    float dA  = __expf(dt * Ah);

    const float* xrow = xBC + (size_t)b * CONV_DIM;
    float xh = xrow[h * HEADDIM + p];
    Bs[p] = xrow[D_INNER + g * D_STATE + p];
    Cs[p] = xrow[D_INNER + NGROUPS * D_STATE + g * D_STATE + p];
    __syncthreads();

    const size_t base = (((size_t)b * NHEADS + h) * HEADDIM + p) * D_STATE;
    const float* sin = ssm_in + base;
    float* sout = ssm_out + base;
    float dtx = dt * xh, acc = 0.0f;
#pragma unroll 8
    for (int n = 0; n < D_STATE; ++n) {
        float s = sin[n] * dA + dtx * Bs[n];
        sout[n] = s;
        acc += s * Cs[n];
    }
    y[(size_t)b * D_INNER + h * HEADDIM + p] = acc + Dvec[h] * xh;
}

// ---------------------------------------------------------------------------
// y = y * silu(z); RMSNorm over D_INNER; * norm_w -> bf16. Block per row.
// ---------------------------------------------------------------------------
__global__ void gate_rmsnorm_kernel(const float* __restrict__ zx,
                                    const float* __restrict__ y,
                                    const float* __restrict__ norm_w,
                                    ushort_t* __restrict__ ybf) {
    __shared__ float red[256];
    const int b = blockIdx.x, t = threadIdx.x;
    const float* z  = zx + (size_t)b * D_IN_PROJ;       // z = first D_INNER cols
    const float* yr = y  + (size_t)b * D_INNER;
    float vals[8], ss = 0.0f;
#pragma unroll
    for (int i = 0; i < 8; ++i) {
        int idx = t + i * 256;
        float v = yr[idx] * siluf_(z[idx]);
        vals[i] = v;
        ss += v * v;
    }
    red[t] = ss;
    __syncthreads();
    for (int s = 128; s > 0; s >>= 1) {
        if (t < s) red[t] += red[t + s];
        __syncthreads();
    }
    float scale = rsqrtf(red[0] * (1.0f / D_INNER) + EPS);
    ushort_t* o = ybf + (size_t)b * D_INNER;
#pragma unroll
    for (int i = 0; i < 8; ++i) {
        int idx = t + i * 256;
        o[idx] = f2bf(vals[i] * scale * norm_w[idx]);
    }
}

// ---------------------------------------------------------------------------
// o1 += gated; o1_bf16 = bf16(o1)
// ---------------------------------------------------------------------------
__global__ void ew_addres_kernel(float* __restrict__ o1,
                                 const float* __restrict__ gated,
                                 ushort_t* __restrict__ o1bf) {
    int i = blockIdx.x * blockDim.x + threadIdx.x;
    float v = o1[i] + gated[i];
    o1[i] = v;
    o1bf[i] = f2bf(v);
}

// ---------------------------------------------------------------------------
// out2 = o1 * sigmoid(g3) -> bf16
// ---------------------------------------------------------------------------
__global__ void ew_outgate_kernel(const float* __restrict__ o1,
                                  const float* __restrict__ g3,
                                  ushort_t* __restrict__ out2bf) {
    int i = blockIdx.x * blockDim.x + threadIdx.x;
    out2bf[i] = f2bf(o1[i] * sigmoidf_(g3[i]));
}

// ---------------------------------------------------------------------------
// Host launch
// ---------------------------------------------------------------------------
extern "C" void kernel_launch(void* const* d_in, const int* in_sizes, int n_in,
                              void* d_out, int out_size, void* d_ws, size_t ws_size,
                              hipStream_t stream) {
    (void)in_sizes; (void)n_in; (void)out_size; (void)ws_size;

    const float* frame_feat   = (const float*)d_in[0];
    const float* conv_state   = (const float*)d_in[1];
    const float* ssm_state    = (const float*)d_in[2];
    const float* tau          = (const float*)d_in[3];
    const float* pre_norm_w   = (const float*)d_in[4];
    const float* pre_norm_b   = (const float*)d_in[5];
    const float* in_gate_w    = (const float*)d_in[6];
    const float* in_gate_b    = (const float*)d_in[7];
    const float* input_proj_w = (const float*)d_in[8];
    const float* input_proj_b = (const float*)d_in[9];
    const float* in_proj_w    = (const float*)d_in[10];
    const float* conv_w       = (const float*)d_in[11];
    const float* conv_b       = (const float*)d_in[12];
    const float* dt_bias      = (const float*)d_in[13];
    const float* A_log        = (const float*)d_in[14];
    const float* Dvec         = (const float*)d_in[15];
    const float* norm_w       = (const float*)d_in[16];
    const float* out_proj_w   = (const float*)d_in[17];
    const float* out_gate_w   = (const float*)d_in[18];
    const float* out_gate_b   = (const float*)d_in[19];
    const float* proj_w       = (const float*)d_in[20];
    const float* proj_b       = (const float*)d_in[21];

    float* clip_out = (float*)d_out;                                  // 512*512
    float* conv_out = clip_out + (size_t)B_SZ * CLIP;                 // 512*2560*4
    float* ssm_out  = conv_out + (size_t)B_SZ * CONV_DIM * D_CONV;    // 512*32*64*64

    // workspace allocator (256B aligned)
    char* ws = (char*)d_ws;
    size_t off = 0;
    auto alloc = [&](size_t bytes) -> void* {
        void* p = ws + off;
        off = (off + bytes + 255) & ~(size_t)255;
        return p;
    };
    ushort_t* Wg_bf   = (ushort_t*)alloc((size_t)HID * IN_DIM * 2);
    ushort_t* Wp_bf   = (ushort_t*)alloc((size_t)HID * IN_DIM * 2);
    ushort_t* Win_bf  = (ushort_t*)alloc((size_t)D_IN_PROJ * HID * 2);
    ushort_t* Wop_bf  = (ushort_t*)alloc((size_t)HID * D_INNER * 2);
    ushort_t* Wog_bf  = (ushort_t*)alloc((size_t)HID * HID * 2);
    ushort_t* Wpr_bf  = (ushort_t*)alloc((size_t)CLIP * HID * 2);
    ushort_t* x_bf    = (ushort_t*)alloc((size_t)B_SZ * IN_DIM * 2);
    float*    g1      = (float*)alloc((size_t)B_SZ * HID * 4);
    float*    g2      = (float*)alloc((size_t)B_SZ * HID * 4);
    float*    gated   = (float*)alloc((size_t)B_SZ * HID * 4);
    ushort_t* gated_bf= (ushort_t*)alloc((size_t)B_SZ * HID * 2);
    float*    zx      = (float*)alloc((size_t)B_SZ * D_IN_PROJ * 4);
    float*    xBC     = (float*)alloc((size_t)B_SZ * CONV_DIM * 4);
    float*    ybuf    = (float*)alloc((size_t)B_SZ * D_INNER * 4);
    ushort_t* y_bf    = (ushort_t*)alloc((size_t)B_SZ * D_INNER * 2);
    float*    o1      = (float*)alloc((size_t)B_SZ * HID * 4);
    ushort_t* o1_bf   = (ushort_t*)alloc((size_t)B_SZ * HID * 2);
    float*    g3      = (float*)alloc((size_t)B_SZ * HID * 4);
    ushort_t* out2_bf = (ushort_t*)alloc((size_t)B_SZ * HID * 2);

    auto cvt = [&](const float* s, ushort_t* d, int n) {
        cvt_bf16_kernel<<<(n + 255) / 256, 256, 0, stream>>>(s, d, n);
    };
    cvt(in_gate_w,    Wg_bf,  HID * IN_DIM);
    cvt(input_proj_w, Wp_bf,  HID * IN_DIM);
    cvt(in_proj_w,    Win_bf, D_IN_PROJ * HID);
    cvt(out_proj_w,   Wop_bf, HID * D_INNER);
    cvt(out_gate_w,   Wog_bf, HID * HID);
    cvt(proj_w,       Wpr_bf, CLIP * HID);

    // 1. pre-LayerNorm -> x (bf16)
    prenorm_kernel<<<B_SZ, 256, 0, stream>>>(frame_feat, pre_norm_w, pre_norm_b, x_bf);

    auto gemm = [&](const ushort_t* A, const ushort_t* W, const float* bias,
                    float* C, int N, int K) {
        dim3 grid((N + 63) / 64, B_SZ / 128);
        gemm_bf16_wmma_kernel<<<grid, 256, 0, stream>>>(A, W, bias, C, N, K);
    };

    // 2. gated = (x@Wp^T + bp) * sigmoid(x@Wg^T + bg)
    gemm(x_bf, Wg_bf, in_gate_b,    g1, HID, IN_DIM);
    gemm(x_bf, Wp_bf, input_proj_b, g2, HID, IN_DIM);
    ew_gated_kernel<<<(B_SZ * HID) / 256, 256, 0, stream>>>(g1, g2, gated, gated_bf);

    // 3. zxbcdt = gated @ in_proj_w^T   (N = 4640)
    gemm(gated_bf, Win_bf, nullptr, zx, D_IN_PROJ, HID);

    // 4. conv state roll + depthwise conv + SiLU
    conv_kernel<<<(B_SZ * CONV_DIM) / 256, 256, 0, stream>>>(
        conv_state, zx, conv_w, conv_b, conv_out, xBC);

    // 5. SSM state update (streams 536 MB of state; HBM-bound floor)
    ssm_kernel<<<dim3(NHEADS, B_SZ), HEADDIM, 0, stream>>>(
        ssm_state, zx, xBC, tau, dt_bias, A_log, Dvec, ssm_out, ybuf);

    // 6. y * silu(z), RMSNorm, * norm_w -> bf16
    gate_rmsnorm_kernel<<<B_SZ, 256, 0, stream>>>(zx, ybuf, norm_w, y_bf);

    // 7. out = y @ out_proj_w^T ; out += gated
    gemm(y_bf, Wop_bf, nullptr, o1, HID, D_INNER);
    ew_addres_kernel<<<(B_SZ * HID) / 256, 256, 0, stream>>>(o1, gated, o1_bf);

    // 8. out *= sigmoid(out @ out_gate_w^T + b)
    gemm(o1_bf, Wog_bf, out_gate_b, g3, HID, HID);
    ew_outgate_kernel<<<(B_SZ * HID) / 256, 256, 0, stream>>>(o1, g3, out2_bf);

    // 9. clip_emb = out @ proj_w^T + proj_b  -> d_out
    gemm(out2_bf, Wpr_bf, proj_b, clip_out, CLIP, HID);
}
